// DVAE_70265664963195
// MI455X (gfx1250) — matrix operand order repaired and checked by reading, not compiled, not run on hardware
//
#include <hip/hip_runtime.h>
#include <hip/hip_bf16.h>
#include <stdint.h>

typedef __attribute__((ext_vector_type(8)))  _Float16 v8h;
typedef __attribute__((ext_vector_type(16))) _Float16 v16h;
typedef __attribute__((ext_vector_type(8)))  float    v8f;

#define B_   512
#define N_   256
#define NVT_ 32
#define HS_  128
#define NZ_  64

__device__ __forceinline__ v8f v8f_zero() {
  v8f r;
  #pragma unroll
  for (int i = 0; i < 8; ++i) r[i] = 0.f;
  return r;
}

// 16 contiguous halves, 32B-aligned
__device__ __forceinline__ v16h ld16h(const _Float16* p) {
  return *(const v16h*)p;
}
// two contiguous 8-half runs (A-fragment pattern), each 16B-aligned
__device__ __forceinline__ v16h ld2x8h(const _Float16* p0, const _Float16* p1) {
  v8h a = *(const v8h*)p0;
  v8h b = *(const v8h*)p1;
  return __builtin_shufflevector(a, b, 0,1,2,3,4,5,6,7,8,9,10,11,12,13,14,15);
}

__device__ __forceinline__ float sigf(float x) {
  return 1.f / (1.f + __expf(-x));
}

// ---------------------------------------------------------------------------
// Pass A: type tables + weight conversions
//   gtab[t][h]   = sigmoid(Wg[h][t]+bg[h]) * Wm[h][t]                (f16)
//   giT[t][g]    = W_ih[g][t] + b_ih[g] + (g<256 ? b_hh[g] : 0)      (f32)
//   Whh16        = (f16)W_hh   [384][128]
//   W1h/W2h      = (f16)W1/W2  [64][128]
// ---------------------------------------------------------------------------
__global__ void __launch_bounds__(256) k_tables(
    const float* Wg, const float* bg, const float* Wm,
    const float* W_ih, const float* b_ih, const float* b_hh,
    const float* W_hh, const float* W1, const float* W2,
    _Float16* gtab, float* giT, _Float16* Whh16, _Float16* W1h, _Float16* W2h)
{
  int i = blockIdx.x * blockDim.x + threadIdx.x;
  if (i < NVT_ * HS_) {
    int t = i / HS_, h = i % HS_;
    float s = sigf(Wg[h * NVT_ + t] + bg[h]);
    gtab[t * HS_ + h] = (_Float16)(s * Wm[h * NVT_ + t]);
  }
  if (i < NVT_ * 3 * HS_) {
    int t = i / (3 * HS_), g = i % (3 * HS_);
    float v = W_ih[g * NVT_ + t] + b_ih[g];
    if (g < 2 * HS_) v += b_hh[g];          // fold b_hh into r/z pre-activations
    giT[t * (3 * HS_) + g] = v;
  }
  if (i < 3 * HS_ * HS_) Whh16[i] = (_Float16)W_hh[i];
  if (i < NZ_ * HS_) { W1h[i] = (_Float16)W1[i]; W2h[i] = (_Float16)W2[i]; }
}

__global__ void __launch_bounds__(256) k_zero(float* p, int n) {
  int i = blockIdx.x * blockDim.x + threadIdx.x;
  if (i < n) p[i] = 0.f;
}

// ---------------------------------------------------------------------------
// Pass B1: GT[b][h][u] = gtab[type[b][u]][h]   (f16, u-contiguous)
// ---------------------------------------------------------------------------
__global__ void __launch_bounds__(256) k_buildGT(
    const int* __restrict__ types, const _Float16* __restrict__ gtab,
    _Float16* __restrict__ GT)
{
  int b = blockIdx.x, u = threadIdx.x;
  int t = types[b * N_ + u];
  const _Float16* src = gtab + t * HS_;
  _Float16* dst = GT + (size_t)b * HS_ * N_ + u;
  #pragma unroll 8
  for (int h = 0; h < HS_; ++h) dst[(size_t)h * N_] = src[h];
}

// ---------------------------------------------------------------------------
// Pass B2: AT[b][v][u] = (f16)adj[b][u][v]   (LDS tile transpose, coalesced)
// ---------------------------------------------------------------------------
__global__ void __launch_bounds__(256) k_buildAT(
    const int* __restrict__ adj, _Float16* __restrict__ AT)
{
  __shared__ _Float16 tile[16][258];          // pad: 258 halves => bank stride 1
  int bs = blockIdx.x;
  int b = bs >> 4, us = (bs & 15) << 4;
  int tid = threadIdx.x;
  #pragma unroll
  for (int r = 0; r < 16; ++r)
    tile[r][tid] = (_Float16)adj[((size_t)b * N_ + us + r) * N_ + tid];
  __syncthreads();
  int ui = tid & 15, vl = tid >> 4;
  #pragma unroll
  for (int r2 = 0; r2 < 16; ++r2) {
    int v = vl * 16 + r2;
    AT[((size_t)b * N_ + v) * N_ + us + ui] = tile[ui][v];
  }
}

// ---------------------------------------------------------------------------
// Pass C (fused): per wave (b, v-tile of 16):
//   1) Hpre tile 16x128 via 64x v_wmma_f32_16x16x32_f16 (K = u)
//   2) stage Hpre f16 in LDS, rebuild A-fragments over K = h
//   3) r/z/n gate matmuls (96 WMMAs) + GRU elementwise + Hg atomic reduce
// ---------------------------------------------------------------------------
__global__ void __launch_bounds__(32) k_main(
    const _Float16* __restrict__ AT, const _Float16* __restrict__ GT,
    const _Float16* __restrict__ Whh16, const float* __restrict__ giT,
    const float* __restrict__ b_hh, const int* __restrict__ types,
    float* __restrict__ Hg)
{
  __shared__ __align__(64) _Float16 Hh[16 * HS_];   // Hpre tile, [v][h]
  const int b = blockIdx.y, vt = blockIdx.x;
  const int lane = threadIdx.x, lo = lane & 15, hi = lane >> 4;
  const int kbA = hi ? 16 : 0;      // A-fragment K base per half-wave
  const int kbB = hi ? 16 : 0;      // B-fragment K base per half-wave

  // ---- Phase 1: Hpre = adjT @ gated -------------------------------------
  v8f accP[8];
  #pragma unroll
  for (int j = 0; j < 8; ++j) accP[j] = v8f_zero();

  const _Float16* Arow  = AT + ((size_t)b * N_ + vt * 16 + lo) * N_;
  const _Float16* Gbase = GT + ((size_t)b * HS_ + lo) * N_;
  #pragma unroll
  for (int kc = 0; kc < 8; ++kc) {                 // K = u, chunks of 32
    v16h a = ld2x8h(Arow + kc * 32 + kbA, Arow + kc * 32 + kbA + 8);
    #pragma unroll
    for (int ht = 0; ht < 8; ++ht) {               // h tiles of 16
      v16h bf = ld16h(Gbase + (size_t)(ht * 16) * N_ + kc * 32 + kbB);
      accP[ht] = __builtin_amdgcn_wmma_f32_16x16x32_f16(
          false, a, false, bf, (short)0, accP[ht], false, false);
    }
  }

  // ---- Phase 2: stage Hpre (f16) to LDS; rebuild A-frags over K=h -------
  #pragma unroll
  for (int j = 0; j < 8; ++j)
    #pragma unroll
    for (int i = 0; i < 8; ++i)
      Hh[(i + 8 * hi) * HS_ + j * 16 + lo] = (_Float16)accP[j][i];
  __syncthreads();   // single-wave WG: NOP barrier, forces LDS ordering

  v16h Af[4];
  #pragma unroll
  for (int kc = 0; kc < 4; ++kc) {                 // K = h, chunks of 32
    const _Float16* p = Hh + lo * HS_ + kc * 32 + kbA;
    Af[kc] = ld2x8h(p, p + 8);
  }

  int tarr[8];
  #pragma unroll
  for (int i = 0; i < 8; ++i)
    tarr[i] = types[b * N_ + vt * 16 + i + 8 * hi];

  // ---- Phase 3: gates + GRU + graph-state reduction ---------------------
  #pragma unroll 2
  for (int j = 0; j < 8; ++j) {                    // output h-tile
    v8f aR = v8f_zero(), aZ = v8f_zero(), aN = v8f_zero();
    const _Float16* wr = Whh16 + (size_t)(j * 16 + lo) * HS_;
    #pragma unroll
    for (int kc = 0; kc < 4; ++kc) {
      int off = kc * 32 + kbB;
      v16h bR = ld16h(wr + off);
      v16h bZ = ld16h(wr + (size_t)(1 * HS_) * HS_ + off);
      v16h bN = ld16h(wr + (size_t)(2 * HS_) * HS_ + off);
      aR = __builtin_amdgcn_wmma_f32_16x16x32_f16(false, Af[kc], false, bR, (short)0, aR, false, false);
      aZ = __builtin_amdgcn_wmma_f32_16x16x32_f16(false, Af[kc], false, bZ, (short)0, aZ, false, false);
      aN = __builtin_amdgcn_wmma_f32_16x16x32_f16(false, Af[kc], false, bN, (short)0, aN, false, false);
    }
    const float bhhn = b_hh[2 * HS_ + j * 16 + lo];  // n-gate b_hh (inside r*gh_n)
    float hg = 0.f;
    #pragma unroll
    for (int i = 0; i < 8; ++i) {
      int n = vt * 16 + i + 8 * hi;
      const float* gi = giT + (size_t)tarr[i] * (3 * HS_) + j * 16 + lo;
      float r  = sigf(gi[0]       + aR[i]);
      float z  = sigf(gi[HS_]     + aZ[i]);
      float nn = tanhf(gi[2*HS_]  + r * (aN[i] + bhhn));
      float hp = accP[j][i];
      float hv = (1.f - z) * nn + z * hp;
      hg += (n >= 1 && n <= N_ - 2) ? hv : 0.f;    // select, keep EXEC uniform
    }
    atomicAdd(Hg + b * HS_ + j * 16 + lo, hg);
  }
}

// ---------------------------------------------------------------------------
// Pass D: mu/logvar = Hg @ W1T/W2T + b   (wave per 16-row tile of B)
// ---------------------------------------------------------------------------
__global__ void __launch_bounds__(32) k_fc(
    const float* __restrict__ Hg, const _Float16* __restrict__ W1h,
    const _Float16* __restrict__ W2h, const float* __restrict__ b1,
    const float* __restrict__ b2, float* __restrict__ out)
{
  const int bt = blockIdx.x;
  const int lane = threadIdx.x, lo = lane & 15, hi = lane >> 4;
  const int kbA = hi ? 16 : 0, kbB = hi ? 16 : 0;

  v8f mu[4], lv[4];
  #pragma unroll
  for (int z = 0; z < 4; ++z) { mu[z] = v8f_zero(); lv[z] = v8f_zero(); }

  const float* hrow = Hg + (size_t)(bt * 16 + lo) * HS_;
  #pragma unroll
  for (int kc = 0; kc < 4; ++kc) {
    v16h a;
    #pragma unroll
    for (int e = 0; e < 8; ++e) {
      a[e]     = (_Float16)hrow[kc * 32 + kbA + e];
      a[e + 8] = (_Float16)hrow[kc * 32 + kbA + 8 + e];
    }
    #pragma unroll
    for (int zt = 0; zt < 4; ++zt) {
      v16h w1 = ld16h(W1h + (size_t)(zt * 16 + lo) * HS_ + kc * 32 + kbB);
      v16h w2 = ld16h(W2h + (size_t)(zt * 16 + lo) * HS_ + kc * 32 + kbB);
      mu[zt] = __builtin_amdgcn_wmma_f32_16x16x32_f16(false, a, false, w1, (short)0, mu[zt], false, false);
      lv[zt] = __builtin_amdgcn_wmma_f32_16x16x32_f16(false, a, false, w2, (short)0, lv[zt], false, false);
    }
  }
  #pragma unroll
  for (int zt = 0; zt < 4; ++zt)
    #pragma unroll
    for (int i = 0; i < 8; ++i) {
      int row = bt * 16 + i + 8 * hi;
      int z   = zt * 16 + lo;
      out[(size_t)row * NZ_ + z]                      = mu[zt][i] + b1[z];
      out[(size_t)B_ * NZ_ + (size_t)row * NZ_ + z]   = lv[zt][i] + b2[z];
    }
}

// ---------------------------------------------------------------------------
extern "C" void kernel_launch(void* const* d_in, const int* in_sizes, int n_in,
                              void* d_out, int out_size, void* d_ws, size_t ws_size,
                              hipStream_t stream)
{
  (void)in_sizes; (void)n_in; (void)out_size; (void)ws_size;
  const int*   node_types = (const int*)  d_in[0];
  const int*   adj        = (const int*)  d_in[1];
  const float* W_ih       = (const float*)d_in[2];
  const float* W_hh       = (const float*)d_in[3];
  const float* b_ih       = (const float*)d_in[4];
  const float* b_hh       = (const float*)d_in[5];
  const float* Wg         = (const float*)d_in[6];
  const float* bg         = (const float*)d_in[7];
  const float* Wm         = (const float*)d_in[8];
  const float* W1         = (const float*)d_in[9];
  const float* b1         = (const float*)d_in[10];
  const float* W2         = (const float*)d_in[11];
  const float* b2         = (const float*)d_in[12];
  float* out = (float*)d_out;

  char* ws = (char*)d_ws;
  size_t off = 0;
  auto take = [&](size_t bytes) -> char* {
    char* p = ws + off;
    off += (bytes + 255) & ~(size_t)255;
    return p;
  };
  _Float16* GT    = (_Float16*)take((size_t)B_ * HS_ * N_ * 2);   // 33.5 MB
  _Float16* AT    = (_Float16*)take((size_t)B_ * N_  * N_ * 2);   // 67.1 MB
  float*    Hg    = (float*)   take((size_t)B_ * HS_ * 4);        // 256 KB
  _Float16* gtab  = (_Float16*)take((size_t)NVT_ * HS_ * 2);
  float*    giT   = (float*)   take((size_t)NVT_ * 3 * HS_ * 4);
  _Float16* Whh16 = (_Float16*)take((size_t)3 * HS_ * HS_ * 2);
  _Float16* W1h   = (_Float16*)take((size_t)NZ_ * HS_ * 2);
  _Float16* W2h   = (_Float16*)take((size_t)NZ_ * HS_ * 2);

  k_tables <<<192, 256, 0, stream>>>(Wg, bg, Wm, W_ih, b_ih, b_hh, W_hh, W1, W2,
                                     gtab, giT, Whh16, W1h, W2h);
  k_zero   <<<(B_ * HS_ + 255) / 256, 256, 0, stream>>>(Hg, B_ * HS_);
  k_buildGT<<<B_, N_, 0, stream>>>(node_types, gtab, GT);
  k_buildAT<<<B_ * 16, 256, 0, stream>>>(adj, AT);
  k_main   <<<dim3(16, B_), 32, 0, stream>>>(AT, GT, Whh16, giT, b_hh, node_types, Hg);
  k_fc     <<<B_ / 16, 32, 0, stream>>>(Hg, W1h, W2h, b1, b2, out);
}
